// attentionHead_22179211117250
// MI455X (gfx1250) — compile-verified, matrix operand order checked by reading, not run on hardware
//
#include <hip/hip_runtime.h>

// ---------------------------------------------------------------------------
// Problem constants (B,S,D,H) = (16,1024,768,12), HD=64
// Strategy: (Q K^T) V  ==  Q (K^T V)   (no softmax in reference!)
// bf16 data path, f32 accumulation via v_wmma_f32_16x16x32_bf16 (wave32).
// GEMM uses async global->LDS double buffering (GLOBAL_LOAD_ASYNC_TO_LDS_B128,
// ASYNCcnt-tracked) when the toolchain exposes the gfx1250 builtins.
// ---------------------------------------------------------------------------

typedef __bf16 bf16;
typedef __attribute__((ext_vector_type(16))) __bf16 v16bf;
typedef __attribute__((ext_vector_type(8)))  __bf16 v8bf;
typedef __attribute__((ext_vector_type(8)))  float  v8f;

#define B_SZ  16
#define S_SZ  1024
#define D_SZ  768
#define H_SZ  12
#define HD_SZ 64
#define BS_SZ (B_SZ * S_SZ)   // 16384 rows

// ---- feature detection: async global->LDS (gfx1250) ----
#if defined(__has_builtin)
#  if __has_builtin(__builtin_amdgcn_global_load_async_to_lds_b128) && \
      __has_builtin(__builtin_amdgcn_s_wait_asynccnt)
#    define USE_ASYNC_LDS 1
#  endif
#endif
#ifndef USE_ASYNC_LDS
#  define USE_ASYNC_LDS 0
#endif

#if USE_ASYNC_LDS
// Builtin prototype (from hipcc diagnostic): pointee is vector_size(16) int,
// src in addrspace(1) (global), dst in addrspace(3) (LDS).
typedef int v4i_ __attribute__((vector_size(16)));
typedef __attribute__((address_space(1))) v4i_* gptr_b128;
typedef __attribute__((address_space(3))) v4i_* lptr_b128;

__device__ __forceinline__ void async_b128(const bf16* g, bf16* l) {
  // per-lane 16-byte global -> LDS copy, tracked by ASYNCcnt.
  // Cast through integers (CK idiom) — generic LDS pointer's low 32 bits are
  // the LDS offset.
  gptr_b128 gp = (gptr_b128)(unsigned long long)g;
  lptr_b128 lp = (lptr_b128)(unsigned int)(unsigned long long)l;
  __builtin_amdgcn_global_load_async_to_lds_b128(gp, lp, 0, 0);
}
template<int N> __device__ __forceinline__ void wait_asynccnt() {
  __builtin_amdgcn_s_wait_asynccnt(N);
}
#endif

__device__ __forceinline__ bf16 f2bf(float f) {
  union { float f; unsigned u; } v; v.f = f;
  unsigned r = v.u + 0x7FFFu + ((v.u >> 16) & 1u);   // round to nearest even
  unsigned short h = (unsigned short)(r >> 16);
  return __builtin_bit_cast(bf16, h);
}

// Load one 16x32 bf16 WMMA A/B fragment from a row-major tile.
// ISA layout: lanes 0-15 row=lo hold K 0-7 and 16-23; lanes 16-31 hold K 8-15, 24-31.
__device__ __forceinline__ v16bf load_frag(const bf16* __restrict__ base, int ld, int lane) {
  const int lo = lane & 15, hi = lane >> 4;
  const bf16* p = base + (size_t)lo * ld + hi * 8;
  v8bf x0 = *(const v8bf*)p;
  v8bf x1 = *(const v8bf*)(p + 16);
  return __builtin_shufflevector(x0, x1, 0,1,2,3,4,5,6,7,8,9,10,11,12,13,14,15);
}

__device__ __forceinline__ v8f wmma_bf16(v16bf a, v16bf b, v8f c) {
  // 8 args: (neg_a, A, neg_b, B, c_mod, C, reuse_a, reuse_b)
  return __builtin_amdgcn_wmma_f32_16x16x32_bf16(false, a, false, b, (short)0, c, false, false);
}

// ---------------------------------------------------------------------------
// Elementwise fp32 -> bf16
// ---------------------------------------------------------------------------
__global__ void cvt_f32_bf16(const float* __restrict__ src, bf16* __restrict__ dst, int n) {
  int i = blockIdx.x * blockDim.x + threadIdx.x;
  int stride = gridDim.x * blockDim.x;
  for (; i < n; i += stride) dst[i] = f2bf(src[i]);
}

// Transpose [rows][cols] fp32 -> [cols][rows] bf16 (weight pre-transpose, one-time)
__global__ void transpose_f32_bf16(const float* __restrict__ src, bf16* __restrict__ dst,
                                   int rows, int cols) {
  int i = blockIdx.x * blockDim.x + threadIdx.x;
  int stride = gridDim.x * blockDim.x;
  const int n = rows * cols;
  for (; i < n; i += stride) {
    int r = i / cols, c = i - r * cols;
    dst[(size_t)c * rows + r] = f2bf(src[i]);
  }
}

// ---------------------------------------------------------------------------
// Tiled bf16 WMMA GEMM:  C[M,N] = A[M,K] * Bt[N,K]^T  (+ bias[n])
// Block tile 128x128, 8 waves (4 x 2), wave tile 32x64 -> 2x4 = 8 wmma accums.
// MODE 0: scatter-store qkv as bf16:  Q[b,h,s,d], Kt[b,h,d,s], Vt[b,h,d,s]
// MODE 1: plain fp32 store (+bias) to Cf
// ---------------------------------------------------------------------------
template<int MODE>
__global__ __launch_bounds__(256)
void gemm_bf16_wmma(const bf16* __restrict__ A, const bf16* __restrict__ Bt,
                    const float* __restrict__ bias,
                    float* __restrict__ Cf,
                    bf16* __restrict__ qb, bf16* __restrict__ ktb, bf16* __restrict__ vtb,
                    int M, int N, int K)
{
  const int t    = threadIdx.x;
  const int lane = t & 31;
  const int w    = t >> 5;
  const int wm   = w & 3;         // wave row  (0..3) -> 32 rows each
  const int wn   = w >> 2;        // wave col  (0..1) -> 64 cols each
  const int lo   = lane & 15, hi = lane >> 4;

  const int bn0 = blockIdx.x * 128;
  const int bm0 = blockIdx.y * 128;

  // cooperative tile load mapping: 256 threads, each owns half a 32-elem row
  const int lr  = t >> 1;          // 0..127
  const int seg = (t & 1) * 16;    // 0 or 16

  const bf16* gA = A  + (size_t)(bm0 + lr) * K + seg;
  const bf16* gB = Bt + (size_t)(bn0 + lr) * K + seg;

  v8f acc[2][4] = {};
  const int KT = K >> 5;           // K chunks of 32

#if USE_ASYNC_LDS
  // ---- async double-buffered pipeline: HBM -> LDS overlapped with WMMA ----
  __shared__ bf16 sA[2][128 * 40];   // 128 rows x 32 k, padded to 40
  __shared__ bf16 sB[2][128 * 40];

  // prologue: kick off tile 0
  async_b128(gA,     &sA[0][lr * 40 + seg]);
  async_b128(gA + 8, &sA[0][lr * 40 + seg + 8]);
  async_b128(gB,     &sB[0][lr * 40 + seg]);
  async_b128(gB + 8, &sB[0][lr * 40 + seg + 8]);

  for (int kt = 0; kt < KT; ++kt) {
    const int cur = kt & 1, nxt = cur ^ 1;
    if (kt + 1 < KT) {
      const bf16* pa = gA + (size_t)(kt + 1) * 32;
      const bf16* pb = gB + (size_t)(kt + 1) * 32;
      async_b128(pa,     &sA[nxt][lr * 40 + seg]);
      async_b128(pa + 8, &sA[nxt][lr * 40 + seg + 8]);
      async_b128(pb,     &sB[nxt][lr * 40 + seg]);
      async_b128(pb + 8, &sB[nxt][lr * 40 + seg + 8]);
      // async loads complete in order: <=4 outstanding => current tile landed
      wait_asynccnt<4>();
    } else {
      wait_asynccnt<0>();
    }
    __syncthreads();               // current tile visible to all waves

    v16bf af[2], bfrag[4];
    #pragma unroll
    for (int i = 0; i < 2; ++i)
      af[i] = load_frag(&sA[cur][(wm * 32 + i * 16) * 40], 40, lane);
    #pragma unroll
    for (int j = 0; j < 4; ++j)
      bfrag[j] = load_frag(&sB[cur][(wn * 64 + j * 16) * 40], 40, lane);

    #pragma unroll
    for (int i = 0; i < 2; ++i)
      #pragma unroll
      for (int j = 0; j < 4; ++j)
        acc[i][j] = wmma_bf16(af[i], bfrag[j], acc[i][j]);

    __syncthreads();               // buffer reuse: reads done before next issue
  }
#else
  // ---- synchronous fallback (VGPR-staged LDS tiles) ----
  __shared__ bf16 sA[128 * 40];
  __shared__ bf16 sB[128 * 40];
  bf16* lA = &sA[lr * 40 + seg];
  bf16* lB = &sB[lr * 40 + seg];

  for (int kt = 0; kt < KT; ++kt) {
    const bf16* pa = gA + (size_t)kt * 32;
    const bf16* pb = gB + (size_t)kt * 32;
    v8bf a0 = *(const v8bf*)pa;
    v8bf a1 = *(const v8bf*)(pa + 8);
    v8bf b0 = *(const v8bf*)pb;
    v8bf b1 = *(const v8bf*)(pb + 8);
    if (kt + 1 < KT) {             // warm L2/L0 for next K-chunk
      __builtin_prefetch(pa + 32, 0, 1);
      __builtin_prefetch(pb + 32, 0, 1);
    }
    __syncthreads();               // previous iteration's LDS reads done
    *(v8bf*)lA = a0; *(v8bf*)(lA + 8) = a1;
    *(v8bf*)lB = b0; *(v8bf*)(lB + 8) = b1;
    __syncthreads();

    v16bf af[2], bfrag[4];
    #pragma unroll
    for (int i = 0; i < 2; ++i)
      af[i] = load_frag(&sA[(wm * 32 + i * 16) * 40], 40, lane);
    #pragma unroll
    for (int j = 0; j < 4; ++j)
      bfrag[j] = load_frag(&sB[(wn * 64 + j * 16) * 40], 40, lane);

    #pragma unroll
    for (int i = 0; i < 2; ++i)
      #pragma unroll
      for (int j = 0; j < 4; ++j)
        acc[i][j] = wmma_bf16(af[i], bfrag[j], acc[i][j]);
  }
#endif

  // Epilogue. C frag layout: c[r] = C[m = r + hi*8][n = lo] of the 16x16 tile.
  #pragma unroll
  for (int i = 0; i < 2; ++i) {
    #pragma unroll
    for (int j = 0; j < 4; ++j) {
      #pragma unroll
      for (int r = 0; r < 8; ++r) {
        const int mg = bm0 + wm * 32 + i * 16 + hi * 8 + r;
        const int ng = bn0 + wn * 64 + j * 16 + lo;
        const float v = acc[i][j][r] + bias[ng];
        if (MODE == 1) {
          Cf[(size_t)mg * N + ng] = v;
        } else {
          const int bb  = mg >> 10;            // / S
          const int ss  = mg & (S_SZ - 1);
          const int sec = ng / D_SZ;           // 0:Q 1:K 2:V
          const int f   = ng - sec * D_SZ;
          const int hh  = f >> 6;
          const int dd  = f & 63;
          const size_t hb = (size_t)(bb * H_SZ + hh);
          const bf16 bv = f2bf(v);
          if (sec == 0)      qb [(hb * S_SZ + ss) * HD_SZ + dd] = bv;  // Q[b,h,s,d]
          else if (sec == 1) ktb[(hb * HD_SZ + dd) * S_SZ + ss] = bv;  // Kt[b,h,d,s]
          else               vtb[(hb * HD_SZ + dd) * S_SZ + ss] = bv;  // Vt[b,h,d,s]
        }
      }
    }
  }
}

// ---------------------------------------------------------------------------
// Attention via associativity: per (b,h):  M = K^T V  (64x64, f32 accum over S),
// M^T staged in LDS as bf16, then O = Q * M over all 1024 rows.
// One workgroup (8 waves) per head.
// ---------------------------------------------------------------------------
__global__ __launch_bounds__(256)
void attn_assoc_wmma(const bf16* __restrict__ q, const bf16* __restrict__ kt,
                     const bf16* __restrict__ vt, bf16* __restrict__ attn)
{
  const int bh = blockIdx.x;
  const int b  = bh / H_SZ;
  const int h  = bh - b * H_SZ;

  const bf16* Qp  = q  + (size_t)bh * S_SZ * HD_SZ;   // [s][d], ld 64
  const bf16* Ktp = kt + (size_t)bh * HD_SZ * S_SZ;   // [dk][s], ld 1024
  const bf16* Vtp = vt + (size_t)bh * HD_SZ * S_SZ;   // [dv][s], ld 1024

  __shared__ bf16 Mt[64 * 72];                        // M^T[dv][dk], padded ld 72

  const int t    = threadIdx.x;
  const int w    = t >> 5;
  const int lane = t & 31;
  const int lo   = lane & 15, hi = lane >> 4;

  // ---- Phase 1: M = K^T V.  16 output tiles (4x4), 2 per wave. ----
  const int tm  = w & 3;            // dk tile
  const int tg  = w >> 2;           // dv tile group (2 tiles)
  const int n0a = (tg * 2) * 16;
  const int n0b = n0a + 16;

  v8f c0 = {}, c1 = {};
  for (int s0 = 0; s0 < S_SZ; s0 += 32) {
    v16bf a  = load_frag(Ktp + (size_t)(tm * 16) * S_SZ + s0, S_SZ, lane);
    v16bf b0 = load_frag(Vtp + (size_t)n0a * S_SZ + s0, S_SZ, lane);
    v16bf b1 = load_frag(Vtp + (size_t)n0b * S_SZ + s0, S_SZ, lane);
    c0 = wmma_bf16(a, b0, c0);
    c1 = wmma_bf16(a, b1, c1);
  }
  // Store transposed (M^T) so phase-2 B fragments read contiguous rows.
  #pragma unroll
  for (int r = 0; r < 8; ++r) {
    const int mdk = tm * 16 + hi * 8 + r;
    Mt[(n0a + lo) * 72 + mdk] = f2bf(c0[r]);
    Mt[(n0b + lo) * 72 + mdk] = f2bf(c1[r]);
  }
  __syncthreads();

  // ---- Phase 2: O = Q * M.  B fragments (M^T rows) are loop-invariant. ----
  v16bf bm[4][2];
  #pragma unroll
  for (int tn = 0; tn < 4; ++tn)
    #pragma unroll
    for (int kk = 0; kk < 2; ++kk)
      bm[tn][kk] = load_frag(&Mt[(tn * 16) * 72 + kk * 32], 72, lane);

  for (int rt = w; rt < S_SZ / 16; rt += 8) {
    const int s0 = rt * 16;
    v16bf a0 = load_frag(Qp + (size_t)s0 * HD_SZ, HD_SZ, lane);
    v16bf a1 = load_frag(Qp + (size_t)s0 * HD_SZ + 32, HD_SZ, lane);
    #pragma unroll
    for (int tn = 0; tn < 4; ++tn) {
      v8f c = {};
      c = wmma_bf16(a0, bm[tn][0], c);
      c = wmma_bf16(a1, bm[tn][1], c);
      #pragma unroll
      for (int r = 0; r < 8; ++r) {
        const int ss  = s0 + hi * 8 + r;
        const int col = h * HD_SZ + tn * 16 + lo;
        attn[(size_t)(b * S_SZ + ss) * D_SZ + col] = f2bf(c[r]);   // [b,s, h*64+dv]
      }
    }
  }
}

// ---------------------------------------------------------------------------
// Launcher
// ---------------------------------------------------------------------------
extern "C" void kernel_launch(void* const* d_in, const int* in_sizes, int n_in,
                              void* d_out, int out_size, void* d_ws, size_t ws_size,
                              hipStream_t stream) {
  (void)in_sizes; (void)n_in; (void)out_size; (void)ws_size;
  const float* x    = (const float*)d_in[0];   // [B,S,D]
  const float* Wqkv = (const float*)d_in[1];   // [D,3D]
  const float* bqkv = (const float*)d_in[2];   // [3D]
  const float* Wout = (const float*)d_in[3];   // [D,D]
  const float* bout = (const float*)d_in[4];   // [D]
  float* out = (float*)d_out;                  // [B,S,D] fp32

  char* ws = (char*)d_ws;
  bf16* xb   = (bf16*)ws;  ws += (size_t)BS_SZ * D_SZ * sizeof(bf16);        // x bf16
  bf16* wqt  = (bf16*)ws;  ws += (size_t)3 * D_SZ * D_SZ * sizeof(bf16);     // Wqkv^T
  bf16* wot  = (bf16*)ws;  ws += (size_t)D_SZ * D_SZ * sizeof(bf16);         // Wout^T
  bf16* qb   = (bf16*)ws;  ws += (size_t)BS_SZ * D_SZ * sizeof(bf16);        // Q[b,h,s,d]
  bf16* ktb  = (bf16*)ws;  ws += (size_t)BS_SZ * D_SZ * sizeof(bf16);        // Kt[b,h,d,s]
  bf16* vtb  = (bf16*)ws;  ws += (size_t)BS_SZ * D_SZ * sizeof(bf16);        // Vt[b,h,d,s]
  bf16* attn = (bf16*)ws;  ws += (size_t)BS_SZ * D_SZ * sizeof(bf16);        // attn out bf16

  cvt_f32_bf16      <<<2048, 256, 0, stream>>>(x, xb, BS_SZ * D_SZ);
  transpose_f32_bf16<<<2048, 256, 0, stream>>>(Wqkv, wqt, D_SZ, 3 * D_SZ);
  transpose_f32_bf16<<<1024, 256, 0, stream>>>(Wout, wot, D_SZ, D_SZ);

  // qkv = x @ Wqkv + b, scattered into Q / Kt / Vt (bf16)
  gemm_bf16_wmma<0><<<dim3((3 * D_SZ) / 128, BS_SZ / 128), 256, 0, stream>>>(
      xb, wqt, bqkv, nullptr, qb, ktb, vtb, BS_SZ, 3 * D_SZ, D_SZ);

  // per-head  O = Q (K^T V)
  attn_assoc_wmma<<<B_SZ * H_SZ, 256, 0, stream>>>(qb, ktb, vtb, attn);

  // out = attn @ Wout + b (fp32)
  gemm_bf16_wmma<1><<<dim3(D_SZ / 128, BS_SZ / 128), 256, 0, stream>>>(
      attn, wot, bout, out, nullptr, nullptr, nullptr, BS_SZ, D_SZ, D_SZ);
}
